// AdaptedTransformerEncoder_56968446214539
// MI455X (gfx1250) — matrix operand order repaired and checked by reading, not compiled, not run on hardware
//
#include <hip/hip_runtime.h>
#include <math.h>

// ---------------------------------------------------------------------------
// MI455X (gfx1250) transformer encoder forward.
//  - all GEMM-like math via v_wmma_f32_16x16x32_bf16 (wave32 WMMA, f32 acc)
//  - weights pre-converted to bf16 once; matrix-consumer activations bf16
//  - Tensor Data Mover (tensor_load_to_lds + s_wait_tensorcnt) for row-major
//    tiles (GEMM A, attention K), with D# LDS padding for aligned 80B rows
//  - double-buffered GEMM, 128x128x32 block tile, 32x64 wave tile
// ---------------------------------------------------------------------------

typedef __attribute__((ext_vector_type(16))) __bf16 v16bf;
typedef __attribute__((ext_vector_type(8)))  float  v8f;
typedef __attribute__((ext_vector_type(4)))  unsigned int u32x4;
typedef __attribute__((ext_vector_type(8)))  int i32x8;
typedef __attribute__((ext_vector_type(4)))  int i32x4;

union FragAB {
    v16bf  v;
    __bf16 h[16];
    uint4  u[2];
};

__device__ __forceinline__ v8f wmma_bf16(v16bf a, v16bf b, v8f c) {
    return __builtin_amdgcn_wmma_f32_16x16x32_bf16(
        false, a, false, b, (short)0, c, false, false);
}

// ---------------------------------------------------------------------------
// Tensor Data Mover: 2D bf16 tile, global -> LDS, optional LDS row padding.
// D# per CDNA5 ISA 8.3/8.4.  6-arg builtin on this toolchain.
// pad codes: interval c -> pad after 2^(c+1) DWORDs; amount a -> (a+1) DWORDs.
// ---------------------------------------------------------------------------
__device__ __forceinline__ void tdm_load_2d_bf16(
    const __bf16* gsrc, unsigned lds_byte_addr,
    unsigned tile_d0, unsigned tile_d1, unsigned stride_elems,
    int pad_en, unsigned pad_interval_c, unsigned pad_amount_m1) {
    unsigned long long ga = (unsigned long long)(uintptr_t)gsrc;
    u32x4 g0;
    g0.x = 1u;                                           // count=1, user desc
    g0.y = lds_byte_addr;                                // LDS byte address
    g0.z = (unsigned)ga;                                 // global addr [31:0]
    g0.w = (unsigned)((ga >> 32) & 0x01FFFFFFull) | (2u << 30);  // type=2
    unsigned w0 = (1u << 16)                             // data_size = 2 bytes
                | (pad_en ? (1u << 20) : 0u)
                | (pad_interval_c << 22)
                | (pad_amount_m1 << 25);
    i32x8 g1;
    g1[0] = (int)w0;
    g1[1] = (int)((tile_d0 & 0xFFFFu) << 16);            // tensor_dim0 lo
    g1[2] = (int)(((tile_d0 >> 16) & 0xFFFFu) | ((tile_d1 & 0xFFFFu) << 16));
    g1[3] = (int)(((tile_d1 >> 16) & 0xFFFFu) | (tile_d0 << 16)); // tile_dim0
    g1[4] = (int)(tile_d1 & 0xFFFFu);                    // tile_dim1
    g1[5] = (int)stride_elems;                           // dim0 stride lo32
    g1[6] = 0;
    g1[7] = 0;
    i32x4 z4 = {0, 0, 0, 0};
    i32x8 z8 = {0, 0, 0, 0, 0, 0, 0, 0};
    __builtin_amdgcn_tensor_load_to_lds(g0, g1, z4, z4, z8, 0);
}

__device__ __forceinline__ unsigned lds_addr_of(const void* p) {
    // generic LDS pointer: [31:0] is the LDS byte offset (ISA aperture rule)
    return (unsigned)(uintptr_t)p;
}

// ---------------------------------------------------------------------------
// fp32 -> bf16 bulk conversion (weights, once per launch)
// ---------------------------------------------------------------------------
__global__ __launch_bounds__(256)
void f32_to_bf16_kernel(const float* __restrict__ src, __bf16* __restrict__ dst,
                        int n) {
    int i = blockIdx.x * 256 + threadIdx.x;
    if (i < n) dst[i] = (__bf16)src[i];
}

// ---------------------------------------------------------------------------
// Positional encoding: h = x + PE (fp32 trunk) and bf16 mirror for GEMM A.
// ---------------------------------------------------------------------------
__global__ __launch_bounds__(256)
void posenc_kernel(const float* __restrict__ x, float* __restrict__ out,
                   __bf16* __restrict__ out_bf, int total) {
    int idx = blockIdx.x * 256 + threadIdx.x;
    if (idx >= total) return;
    int d = idx % 768;
    int s = (idx / 768) & 1023;
    float e   = (float)(d & ~1) * (-9.210340371976184f / 768.0f);
    float a   = (float)s * __expf(e);
    float pe  = (d & 1) ? __cosf(a) : __sinf(a);
    float v   = x[idx] + pe;
    out[idx]    = v;
    out_bf[idx] = (__bf16)v;
}

// ---------------------------------------------------------------------------
// GEMM: C[M,N] = act(A[M,K]bf16 @ W[K,N]bf16 + bias[N]f32)
// 128x128x32 block tile, 8 waves (4x2), wave tile 32x64: 8 WMMA / k-step,
// each B fragment reused by 2 WMMAs, each A fragment by 4.
// A tile: TDM -> As[buf][m][40] (padded 80B rows).  B tile: manual transpose
// staging -> Bs[buf][n][40].  Double-buffered; wave0 owns TENSORcnt.
// ---------------------------------------------------------------------------
template <bool RELU, bool OUTBF>
__global__ __launch_bounds__(256)
void gemm_bf16_wmma(const __bf16* __restrict__ A, const __bf16* __restrict__ W,
                    const float* __restrict__ bias, void* __restrict__ Cout,
                    int M, int N, int K) {
    __shared__ __bf16 As[2][128][40];    // 80B rows via TDM pad (16B aligned)
    __shared__ __bf16 Bs[2][128][40];

    const int tid  = threadIdx.x;
    const int wid  = tid >> 5;
    const int lane = tid & 31;
    const int hrow = lane & 15;
    const int half = lane >> 4;

    const int m0 = blockIdx.y * 128;
    const int n0 = blockIdx.x * 128;
    const int wm = wid & 3;              // 0..3 -> 32-row strip
    const int wn = wid >> 2;             // 0..1 -> 64-col strip

    v8f acc[2][4] = {};

    auto stage_b = [&](int buf, int kb) {
#pragma unroll
        for (int i = 0; i < 2; ++i) {
            int slot = tid + i * 256;          // 0..511 = 32 k x 16 col-groups
            int kr   = slot >> 4;              // 0..31
            int c8   = (slot & 15) << 3;       // 0..120
            union { uint4 u; __bf16 h[8]; } t;
            t.u = *(const uint4*)(W + (size_t)(kb + kr) * N + n0 + c8);
#pragma unroll
            for (int j = 0; j < 8; ++j) Bs[buf][c8 + j][kr] = t.h[j];
        }
    };
    auto issue_a = [&](int buf, int kb) {
        // tile 128 rows x 32 k; 64B rows padded +16B -> stride 40 elements
        tdm_load_2d_bf16(A + (size_t)m0 * K + kb, lds_addr_of(&As[buf][0][0]),
                         /*d0=*/32, /*d1=*/128, /*stride=*/(unsigned)K,
                         /*pad=*/1, /*interval 64B=*/3, /*amount 16B=*/3);
    };

    const int NKB = K >> 5;
    stage_b(0, 0);
    if (wid == 0) {
        issue_a(0, 0);
        __builtin_amdgcn_s_wait_tensorcnt(0);
    }
    __syncthreads();

    for (int t = 0; t < NKB; ++t) {
        const int cur = t & 1;
        if (t + 1 < NKB) {
            stage_b(cur ^ 1, (t + 1) << 5);
            if (wid == 0) issue_a(cur ^ 1, (t + 1) << 5);
        }
        // ---- compute on current buffers ----
        FragAB fa[2];
        const int akb = half << 3;
#pragma unroll
        for (int mi = 0; mi < 2; ++mi) {
            const int arow = wm * 32 + mi * 16 + hrow;
            fa[mi].u[0] = *(const uint4*)&As[cur][arow][akb];
            fa[mi].u[1] = *(const uint4*)&As[cur][arow][akb + 16];
        }
#pragma unroll
        for (int nt = 0; nt < 4; ++nt) {
            FragAB fb;
            const int col = wn * 64 + nt * 16 + hrow;
            const int bkb = half << 3;
            fb.u[0] = *(const uint4*)&Bs[cur][col][bkb];
            fb.u[1] = *(const uint4*)&Bs[cur][col][bkb + 16];
#pragma unroll
            for (int mi = 0; mi < 2; ++mi)
                acc[mi][nt] = wmma_bf16(fa[mi].v, fb.v, acc[mi][nt]);
        }
        __syncthreads();                       // cur reads done, nxt B staged
        if (wid == 0) __builtin_amdgcn_s_wait_tensorcnt(0);
        __syncthreads();                       // nxt A tile landed
    }

#pragma unroll
    for (int nt = 0; nt < 4; ++nt) {
        int col  = n0 + wn * 64 + nt * 16 + hrow;
        float bv = bias[col];
#pragma unroll
        for (int mi = 0; mi < 2; ++mi) {
#pragma unroll
            for (int r = 0; r < 8; ++r) {
                int row = m0 + wm * 32 + mi * 16 + r + (half << 3);
                float v = acc[mi][nt][r] + bv;
                if (RELU) v = fmaxf(v, 0.0f);
                if (OUTBF)
                    ((__bf16*)Cout)[(size_t)row * N + col] = (__bf16)v;
                else
                    ((float*)Cout)[(size_t)row * N + col] = v;
            }
        }
    }
}

// ---------------------------------------------------------------------------
// Flash attention over bf16 qkv [B*S, 2304] (q|k|v per head at h*64+d).
// 128 threads (4 waves), wave = 16-query tile, 32-key streaming blocks.
// K tile: TDM row-major [key][d] (144B padded rows).  V tile: manual
// transposed staging [d][key].  P transposed through per-wave LDS.
// ---------------------------------------------------------------------------
__global__ __launch_bounds__(128)
void flash_attn_kernel(const __bf16* __restrict__ qkv, __bf16* __restrict__ out,
                       float scale) {
    constexpr int S = 1024, D3 = 2304;
    __shared__ __bf16 Kl[32][72];      // TDM target: 128B rows + 16B pad
    __shared__ __bf16 Vl[64][40];      // transposed [d][key]
    __shared__ __bf16 Pl[4][16][40];

    const int tid  = threadIdx.x;
    const int w    = tid >> 5;
    const int lane = tid & 31;
    const int hrow = lane & 15;
    const int half = lane >> 4;

    const int b    = blockIdx.z;
    const int head = blockIdx.y;
    const int qs   = blockIdx.x * 64 + w * 16;

    FragAB fq[2];
    {
        const int q = qs + hrow;
        const __bf16* qrow = qkv + (size_t)(b * S + q) * D3 + head * 64;
#pragma unroll
        for (int kk = 0; kk < 2; ++kk) {
            int dbase = kk * 32 + (half << 3);
#pragma unroll
            for (int j = 0; j < 8; ++j) {
                fq[kk].h[j]     = (__bf16)((float)qrow[dbase + j] * scale);
                fq[kk].h[8 + j] = (__bf16)((float)qrow[dbase + 16 + j] * scale);
            }
        }
    }

    v8f o[4] = {};
    float mrow[8], lrow[8];
#pragma unroll
    for (int r = 0; r < 8; ++r) { mrow[r] = -INFINITY; lrow[r] = 0.0f; }

    for (int kb = 0; kb < S; kb += 32) {
        __syncthreads();               // previous block's reads complete
        if (w == 0) {
            // K tile: 32 keys x 64 d, row stride D3; pad 128B rows +16B
            tdm_load_2d_bf16(qkv + (size_t)(b * S + kb) * D3 + 768 + head * 64,
                             lds_addr_of(&Kl[0][0]),
                             /*d0=*/64, /*d1=*/32, /*stride=*/D3,
                             /*pad=*/1, /*interval 128B=*/4, /*amount 16B=*/3);
        }
        // V tile staged transposed by all threads
#pragma unroll
        for (int i = 0; i < 2; ++i) {
            int slot = tid + i * 128;       // 0..255 = 32 keys x 8 col-groups
            int key  = slot >> 3;
            int c8   = (slot & 7) << 3;
            union { uint4 u; __bf16 h[8]; } t;
            t.u = *(const uint4*)(qkv + (size_t)(b * S + kb + key) * D3 +
                                  1536 + head * 64 + c8);
#pragma unroll
            for (int j = 0; j < 8; ++j) Vl[c8 + j][key] = t.h[j];
        }
        if (w == 0) __builtin_amdgcn_s_wait_tensorcnt(0);
        __syncthreads();

        // scores: Q(16x64) @ K^T(64x32)
        v8f sc[2] = {};
#pragma unroll
        for (int nt = 0; nt < 2; ++nt) {
#pragma unroll
            for (int kk = 0; kk < 2; ++kk) {
                FragAB fk;
                const int key = nt * 16 + hrow;
                const int db  = kk * 32 + (half << 3);
                fk.u[0] = *(const uint4*)&Kl[key][db];
                fk.u[1] = *(const uint4*)&Kl[key][db + 16];
                sc[nt] = wmma_bf16(fq[kk].v, fk.v, sc[nt]);
            }
        }

        // online softmax (rows live in 16-lane halves)
#pragma unroll
        for (int r = 0; r < 8; ++r) {
            float mx = fmaxf(sc[0][r], sc[1][r]);
            mx = fmaxf(mx, __shfl_xor(mx, 1));
            mx = fmaxf(mx, __shfl_xor(mx, 2));
            mx = fmaxf(mx, __shfl_xor(mx, 4));
            mx = fmaxf(mx, __shfl_xor(mx, 8));
            float mnew = fmaxf(mrow[r], mx);
            float corr = __expf(mrow[r] - mnew);
            float p0 = __expf(sc[0][r] - mnew);
            float p1 = __expf(sc[1][r] - mnew);
            float ps = p0 + p1;
            ps += __shfl_xor(ps, 1);
            ps += __shfl_xor(ps, 2);
            ps += __shfl_xor(ps, 4);
            ps += __shfl_xor(ps, 8);
            lrow[r] = lrow[r] * corr + ps;
            mrow[r] = mnew;
#pragma unroll
            for (int nt = 0; nt < 4; ++nt) o[nt][r] *= corr;
            int prow = r + (half << 3);
            Pl[w][prow][hrow]      = (__bf16)p0;
            Pl[w][prow][16 + hrow] = (__bf16)p1;
        }
        asm volatile("s_wait_dscnt 0x0" ::: "memory");

        // O += P(16x32) @ V(32x64)
        FragAB fp;
        fp.u[0] = *(const uint4*)&Pl[w][hrow][half << 3];
        fp.u[1] = *(const uint4*)&Pl[w][hrow][(half << 3) + 16];
#pragma unroll
        for (int nt = 0; nt < 4; ++nt) {
            FragAB fv;
            const int dcol = nt * 16 + hrow;
            const int kk2  = half << 3;
            fv.u[0] = *(const uint4*)&Vl[dcol][kk2];
            fv.u[1] = *(const uint4*)&Vl[dcol][kk2 + 16];
            o[nt] = wmma_bf16(fp.v, fv.v, o[nt]);
        }
    }

#pragma unroll
    for (int nt = 0; nt < 4; ++nt) {
#pragma unroll
        for (int r = 0; r < 8; ++r) {
            int row  = qs + r + (half << 3);
            int dcol = nt * 16 + hrow;
            out[(size_t)(b * S + row) * 768 + head * 64 + dcol] =
                (__bf16)(o[nt][r] / lrow[r]);
        }
    }
}

// ---------------------------------------------------------------------------
// Residual add + LayerNorm (D=768), fp32 trunk + optional bf16 mirror.
// ---------------------------------------------------------------------------
__global__ __launch_bounds__(256)
void add_ln_kernel(const float* __restrict__ x, const float* __restrict__ res,
                   const float* __restrict__ g, const float* __restrict__ bb,
                   float* __restrict__ out, __bf16* __restrict__ out_bf) {
    __shared__ float r1[8], r2[8];
    const int row = blockIdx.x;
    const float* xr = x + (size_t)row * 768;
    const float* rr = res ? res + (size_t)row * 768 : nullptr;
    float v[3];
    float s = 0.0f, s2 = 0.0f;
#pragma unroll
    for (int i = 0; i < 3; ++i) {
        int d = threadIdx.x + i * 256;
        float t = xr[d] + (rr ? rr[d] : 0.0f);
        v[i] = t;
        s  += t;
        s2 += t * t;
    }
#pragma unroll
    for (int off = 16; off >= 1; off >>= 1) {
        s  += __shfl_xor(s, off);
        s2 += __shfl_xor(s2, off);
    }
    const int wid = threadIdx.x >> 5;
    if ((threadIdx.x & 31) == 0) { r1[wid] = s; r2[wid] = s2; }
    __syncthreads();
    s = 0.0f; s2 = 0.0f;
#pragma unroll
    for (int wI = 0; wI < 8; ++wI) { s += r1[wI]; s2 += r2[wI]; }
    float mu   = s * (1.0f / 768.0f);
    float var  = s2 * (1.0f / 768.0f) - mu * mu;
    float rstd = rsqrtf(var + 1e-5f);
#pragma unroll
    for (int i = 0; i < 3; ++i) {
        int d = threadIdx.x + i * 256;
        float o = (v[i] - mu) * rstd * g[d] + bb[d];
        out[(size_t)row * 768 + d] = o;
        if (out_bf) out_bf[(size_t)row * 768 + d] = (__bf16)o;
    }
}

// ---------------------------------------------------------------------------
// Orchestration
// ---------------------------------------------------------------------------
extern "C" void kernel_launch(void* const* d_in, const int* in_sizes, int n_in,
                              void* d_out, int out_size, void* d_ws, size_t ws_size,
                              hipStream_t stream) {
    (void)in_sizes; (void)n_in; (void)out_size; (void)ws_size;

    constexpr int B = 4, S = 1024, D = 768, F = 3072, L = 6, H = 12;
    constexpr int M = B * S;

    const float* x     = (const float*)d_in[0];
    const float* Wqkv  = (const float*)d_in[1];
    const float* bqkv  = (const float*)d_in[2];
    const float* Wo    = (const float*)d_in[3];
    const float* bo    = (const float*)d_in[4];
    const float* ln1_g = (const float*)d_in[5];
    const float* ln1_b = (const float*)d_in[6];
    const float* W1    = (const float*)d_in[7];
    const float* b1    = (const float*)d_in[8];
    const float* W2    = (const float*)d_in[9];
    const float* b2    = (const float*)d_in[10];
    const float* ln2_g = (const float*)d_in[11];
    const float* ln2_b = (const float*)d_in[12];
    const float* lnf_g = (const float*)d_in[13];
    const float* lnf_b = (const float*)d_in[14];

    // workspace: fp32 trunk first, then bf16 region
    float* wsf  = (float*)d_ws;
    float* h    = wsf;  wsf += (size_t)M * D;          // 12 MB
    float* proj = wsf;  wsf += (size_t)M * D;          // 12 MB
    __bf16* wsb   = (__bf16*)wsf;
    __bf16* h_bf  = wsb;  wsb += (size_t)M * D;        // 6 MB
    __bf16* qkv   = wsb;  wsb += (size_t)M * 3 * D;    // 18 MB
    __bf16* attn  = wsb;  wsb += (size_t)M * D;        // 6 MB
    __bf16* ffn   = wsb;  wsb += (size_t)M * F;        // 24 MB
    __bf16* wqkvb = wsb;  wsb += (size_t)L * D * 3 * D;
    __bf16* wob   = wsb;  wsb += (size_t)L * D * D;
    __bf16* w1b   = wsb;  wsb += (size_t)L * D * F;
    __bf16* w2b   = wsb;  wsb += (size_t)L * F * D;    // weights ~81 MB bf16

    auto conv = [&](const float* s, __bf16* d, size_t n) {
        f32_to_bf16_kernel<<<((int)n + 255) / 256, 256, 0, stream>>>(s, d, (int)n);
    };
    conv(Wqkv, wqkvb, (size_t)L * D * 3 * D);
    conv(Wo,   wob,   (size_t)L * D * D);
    conv(W1,   w1b,   (size_t)L * D * F);
    conv(W2,   w2b,   (size_t)L * F * D);

    const int total = M * D;
    posenc_kernel<<<(total + 255) / 256, 256, 0, stream>>>(x, h, h_bf, total);

    const float scale = 0.125f;   // 1/sqrt(64)

    for (int l = 0; l < L; ++l) {
        const __bf16* wqkv_l = wqkvb + (size_t)l * D * 3 * D;
        const __bf16* wo_l   = wob   + (size_t)l * D * D;
        const __bf16* w1_l   = w1b   + (size_t)l * D * F;
        const __bf16* w2_l   = w2b   + (size_t)l * F * D;

        // QKV projection -> bf16 qkv
        gemm_bf16_wmma<false, true><<<dim3(3 * D / 128, M / 128), 256, 0, stream>>>(
            h_bf, wqkv_l, bqkv + (size_t)l * 3 * D, qkv, M, 3 * D, D);

        flash_attn_kernel<<<dim3(S / 64, H, B), 128, 0, stream>>>(qkv, attn, scale);

        // output projection -> fp32 proj (residual consumer)
        gemm_bf16_wmma<false, false><<<dim3(D / 128, M / 128), 256, 0, stream>>>(
            attn, wo_l, bo + (size_t)l * D, proj, M, D, D);

        add_ln_kernel<<<M, 256, 0, stream>>>(h, proj, ln1_g + (size_t)l * D,
                                             ln1_b + (size_t)l * D, h, h_bf);

        // FFN1 + ReLU -> bf16 ffn
        gemm_bf16_wmma<true, true><<<dim3(F / 128, M / 128), 256, 0, stream>>>(
            h_bf, w1_l, b1 + (size_t)l * F, ffn, M, F, D);

        // FFN2 -> fp32 proj
        gemm_bf16_wmma<false, false><<<dim3(D / 128, M / 128), 256, 0, stream>>>(
            ffn, w2_l, b2 + (size_t)l * D, proj, M, D, F);

        add_ln_kernel<<<M, 256, 0, stream>>>(h, proj, ln2_g + (size_t)l * D,
                                             ln2_b + (size_t)l * D, h, h_bf);
    }

    add_ln_kernel<<<M, 256, 0, stream>>>(h, nullptr, lnf_g, lnf_b,
                                         (float*)d_out, nullptr);
}
